// GCNResidual_old_4329327034523
// MI455X (gfx1250) — matrix operand order repaired and checked by reading, not compiled, not run on hardware
//
#include <hip/hip_runtime.h>

#define N_NODES 100000
#define N_EDGES 1000000
#define F_INDIM 128
#define HDIM    256
#define NGRAPH  64
#define EPSB    1e-5f

typedef __attribute__((ext_vector_type(16))) _Float16 v16h;
typedef __attribute__((ext_vector_type(8)))  _Float16 v8h;
typedef __attribute__((ext_vector_type(4)))  _Float16 v4h;
typedef __attribute__((ext_vector_type(8)))  float    v8f;
typedef __attribute__((ext_vector_type(4)))  float    f4;
typedef __attribute__((ext_vector_type(4)))  int      v4i;

#if defined(__has_builtin)
#if __has_builtin(__builtin_amdgcn_global_load_async_to_lds_b128)
#define HAVE_ASYNC_LDS 1
#endif
#if __has_builtin(__builtin_amdgcn_s_wait_asynccnt)
#define HAVE_WAIT_ASYNC 1
#endif
#endif

#if HAVE_ASYNC_LDS
typedef __attribute__((address_space(1))) v4i as1_v4i;   // global int4
typedef __attribute__((address_space(3))) v4i as3_v4i;   // LDS int4
#endif

// ---------------- utility ----------------
__global__ void k_zero(float* __restrict__ p, int n) {
    int i = blockIdx.x * blockDim.x + threadIdx.x;
    if (i < n) p[i] = 0.f;
}

// ---------------- gcn_norm ----------------
__global__ void k_deg(const int* __restrict__ dst, float* __restrict__ deg) {
    int e = blockIdx.x * blockDim.x + threadIdx.x;
    if (e < N_EDGES) atomicAdd(&deg[dst[e]], 1.f);
}

__global__ void k_degfin(const float* __restrict__ deg,
                         float* __restrict__ dis, float* __restrict__ selfw) {
    int n = blockIdx.x * blockDim.x + threadIdx.x;
    if (n < N_NODES) {
        float d = deg[n] + 1.f;
        dis[n]   = rsqrtf(d);
        selfw[n] = 1.f / d;
    }
}

__global__ void k_enorm(const int* __restrict__ src, const int* __restrict__ dst,
                        const float* __restrict__ dis, float* __restrict__ enorm) {
    int e = blockIdx.x * blockDim.x + threadIdx.x;
    if (e < N_EDGES) enorm[e] = dis[src[e]] * dis[dst[e]];
}

// ---------------- weight pre-pack: f32 [K x 256] -> f16 fragment-major ----------------
// Layout: Wp[(((kt32*16 + nt)*32 + lane)*16 + h], lane's 16 halves contiguous (32 B),
// so a B fragment is two coalesced global_load_b128 per lane.
// Mapping (ISA 16-bit B 32x16): k = kt32*32 + (h>>3)*16 + (lane>>4)*8 + (h&7), col = nt*16 + (lane&15).
__global__ void k_packW(const float* __restrict__ W, _Float16* __restrict__ Wp, int KH) {
    int i = blockIdx.x * blockDim.x + threadIdx.x;
    if (i >= KH) return;
    int h    = i & 15;
    int lane = (i >> 4) & 31;
    int rest = i >> 9;
    int nt   = rest & 15;          // HDIM/16 == 16
    int kt32 = rest >> 4;
    int krow = (kt32 << 5) + ((h >> 3) << 4) + ((lane >> 4) << 3) + (h & 7);
    int ncol = (nt << 4) + (lane & 15);
    Wp[i] = (_Float16)W[(size_t)krow * HDIM + ncol];
}

// ---------------- WMMA GEMM: C[M x 256] = A[M x K](f32) * Wpacked(f16) ----------------
// 128 threads = 4 waves; block owns 16 rows x 256 cols; wave w owns cols [64w, 64w+64).
// A tile (16x32 f32, 2 KB) staged via async-load-to-LDS with double buffering;
// f32->f16 conversion happens in the fragment build (co-executes with WMMA).
template <int KDIM>
__global__ __launch_bounds__(128)
void k_gemm_wmma(const float* __restrict__ A, const _Float16* __restrict__ Wp,
                 float* __restrict__ C) {
    // 36-dword row stride: 16B-aligned rows, banks m*36 mod 64 all distinct.
    __shared__ __attribute__((aligned(16))) float As32[2][16][36];
    const int tid  = threadIdx.x;
    const int lane = tid & 31;
    const int wave = tid >> 5;
    const int m0   = blockIdx.x * 16;
    const int n0   = wave * 64;

    const int sr  = tid >> 3;          // staging row 0..15
    const int sk4 = (tid & 7) * 4;     // staging k offset (4 floats = 16 B per thread)

#if HAVE_ASYNC_LDS
#define STAGE_A(buf, kt)                                                              \
    do {                                                                              \
        const float* gp_ = &A[(size_t)(m0 + sr) * KDIM + (kt) + sk4];                 \
        __builtin_amdgcn_global_load_async_to_lds_b128(                               \
            (as1_v4i*)(unsigned long long)(const void*)gp_,                           \
            (as3_v4i*)(unsigned int)(unsigned long long)                              \
                (const void*)&As32[(buf)][sr][sk4],                                   \
            0, 0);                                                                    \
    } while (0)
#if HAVE_WAIT_ASYNC
#define WAITA(n) __builtin_amdgcn_s_wait_asynccnt(n)
#else
#define WAITA(n) asm volatile("s_wait_asynccnt %0" ::"i"(n) : "memory")
#endif
#else
#define STAGE_A(buf, kt)                                                              \
    do {                                                                              \
        *(f4*)&As32[(buf)][sr][sk4] =                                                 \
            *(const f4*)&A[(size_t)(m0 + sr) * KDIM + (kt) + sk4];                    \
    } while (0)
#define WAITA(n) do {} while (0)
#endif

    v8f acc[4];
    #pragma unroll
    for (int j = 0; j < 4; ++j) acc[j] = v8f{};

    const int mrow = lane & 15;
    const int kbA  = (lane >> 4) * 8;   // A frag: lanes 0-15 own K 0-7/16-23, lanes 16-31 own 8-15/24-31

    const int ktmax = KDIM / 32;
    STAGE_A(0, 0);

    for (int kt32 = 0; kt32 < ktmax; ++kt32) {
        const int cur = kt32 & 1;
        if (kt32 + 1 < ktmax) {
            STAGE_A(cur ^ 1, (kt32 + 1) * 32);
            WAITA(1);                       // current tile's copies complete
        } else {
            WAITA(0);
        }
        __syncthreads();                    // tile visible to all 4 waves

        // A fragment: four 16B LDS reads + packed cvt  (ISA 16-bit A 16x32 layout)
        union { v16h v; v4h q[4]; } af;
        {
            const f4* a0 = (const f4*)&As32[cur][mrow][kbA];
            const f4* a1 = (const f4*)&As32[cur][mrow][16 + kbA];
            f4 x0 = a0[0], x1 = a0[1], x2 = a1[0], x3 = a1[1];
            #pragma unroll
            for (int i = 0; i < 4; ++i) {
                af.q[0][i] = (_Float16)x0[i];
                af.q[1][i] = (_Float16)x1[i];
                af.q[2][i] = (_Float16)x2[i];
                af.q[3][i] = (_Float16)x3[i];
            }
        }

        // B fragments from packed f16 weights: two global_load_b128 each
        #pragma unroll
        for (int j = 0; j < 4; ++j) {
            const int nt = wave * 4 + j;
            const v8h* wp =
                (const v8h*)&Wp[(((size_t)kt32 * 16 + nt) * 32 + lane) * 16];
            union { v16h v; v8h h2[2]; } bf;
            bf.h2[0] = wp[0];
            bf.h2[1] = wp[1];
            acc[j] = __builtin_amdgcn_wmma_f32_16x16x32_f16(
                         false, af.v, false, bf.v, (short)0, acc[j], false, false);
        }
        __syncthreads();                    // all waves done reading before next overwrite
    }

    // D layout: VGPR v -> row m0 + v + 8*(lane>=16), col = lane&15
    const int rbase = m0 + 8 * (lane >> 4);
    #pragma unroll
    for (int j = 0; j < 4; ++j)
        #pragma unroll
        for (int v = 0; v < 8; ++v)
            C[(size_t)(rbase + v) * HDIM + n0 + j * 16 + (lane & 15)] = acc[j][v];
#undef STAGE_A
#undef WAITA
}

// ---------------- aggregation ----------------
// agg = hw * self_w + b   (float4, full overwrite -> no pre-zero needed)
__global__ void k_agg_init(const float* __restrict__ hw, const float* __restrict__ selfw,
                           const float* __restrict__ bias, float* __restrict__ agg) {
    int i = blockIdx.x * blockDim.x + threadIdx.x;     // over N*H/4
    if (i >= N_NODES * (HDIM / 4)) return;
    int n = i >> 6;            // H/4 == 64
    int c = i & 63;
    f4 h = ((const f4*)hw)[i];
    f4 b = ((const f4*)bias)[c];
    ((f4*)agg)[i] = h * selfw[n] + b;
}

// edge scatter: 64 threads per edge, float4 gather + 4x f32 atomics (L2-resident)
__global__ void k_scatter(const int* __restrict__ src, const int* __restrict__ dst,
                          const float* __restrict__ enorm, const float* __restrict__ hw,
                          float* __restrict__ agg) {
    int t = blockIdx.x * blockDim.x + threadIdx.x;     // E*64 = 64M < 2^31
    if (t >= N_EDGES * 64) return;
    int e = t >> 6;
    int c = t & 63;
    int s = src[e], d = dst[e];
    float w = enorm[e];
    f4 v = ((const f4*)hw)[s * 64 + c] * w;
    float* base = &agg[(size_t)d * HDIM + c * 4];
    atomicAdd(base + 0, v.x);
    atomicAdd(base + 1, v.y);
    atomicAdd(base + 2, v.z);
    atomicAdd(base + 3, v.w);
}

// ---------------- batchnorm ----------------
__global__ void k_bnstats(const float* __restrict__ agg,
                          float* __restrict__ cs, float* __restrict__ cs2) {
    int c = threadIdx.x;               // one column per thread, blockDim == HDIM
    float s = 0.f, s2 = 0.f;
    for (int r = blockIdx.x; r < N_NODES; r += gridDim.x) {
        float v = agg[(size_t)r * HDIM + c];
        s += v; s2 += v * v;
    }
    atomicAdd(&cs[c], s);
    atomicAdd(&cs2[c], s2);
}

__global__ void k_bnfin(const float* __restrict__ cs, const float* __restrict__ cs2,
                        const float* __restrict__ g, const float* __restrict__ be,
                        float* __restrict__ scale, float* __restrict__ shift) {
    int c = threadIdx.x;
    const float invN = 1.f / (float)N_NODES;
    float m  = cs[c] * invN;
    float v  = cs2[c] * invN - m * m;
    float iv = rsqrtf(v + EPSB);
    float sc = iv * g[c];
    scale[c] = sc;
    shift[c] = be[c] - m * sc;
}

// X = relu(agg*scale + shift) + X   (fused BN-affine + ReLU + in-place residual)
__global__ void k_apply(const float* __restrict__ agg, const float* __restrict__ scale,
                        const float* __restrict__ shift, float* __restrict__ X) {
    int i = blockIdx.x * blockDim.x + threadIdx.x;     // over N*H/4
    if (i >= N_NODES * (HDIM / 4)) return;
    int c = i & 63;
    f4 a = ((const f4*)agg)[i] * ((const f4*)scale)[c] + ((const f4*)shift)[c];
    a.x = fmaxf(a.x, 0.f); a.y = fmaxf(a.y, 0.f);
    a.z = fmaxf(a.z, 0.f); a.w = fmaxf(a.w, 0.f);
    ((f4*)X)[i] = a + ((const f4*)X)[i];
}

// ---------------- pool + head ----------------
__global__ void k_pool(const int* __restrict__ batch, const float* __restrict__ X,
                       float* __restrict__ gsum, float* __restrict__ gcnt) {
    int n = blockIdx.x;
    int c = threadIdx.x;
    int g = batch[n];
    atomicAdd(&gsum[g * HDIM + c], X[(size_t)n * HDIM + c]);
    if (c == 0) atomicAdd(&gcnt[g], 1.f);
}

__global__ void k_final(const float* __restrict__ gsum, const float* __restrict__ gcnt,
                        const float* __restrict__ Wl, const float* __restrict__ bl,
                        float* __restrict__ out) {
    __shared__ float red[HDIM];
    int g = blockIdx.x, c = threadIdx.x;
    float cnt = fmaxf(gcnt[g], 1.f);
    red[c] = (gsum[g * HDIM + c] / cnt) * Wl[c];
    __syncthreads();
    #pragma unroll
    for (int s = HDIM / 2; s > 0; s >>= 1) {
        if (c < s) red[c] += red[c + s];
        __syncthreads();
    }
    if (c == 0) out[g] = red[0] + bl[0];
}

// ---------------- host-side block driver ----------------
static void run_block(hipStream_t stream,
                      const float* hw, float* S, float* X,
                      const float* selfw, const int* src, const int* dst,
                      const float* enorm, const float* bias,
                      const float* g, const float* be,
                      float* cs, float* cs2, float* scale, float* shift) {
    const int NH4 = N_NODES * (HDIM / 4);
    k_agg_init<<<(NH4 + 255) / 256, 256, 0, stream>>>(hw, selfw, bias, S);
    k_scatter<<<(N_EDGES * 64 + 255) / 256, 256, 0, stream>>>(src, dst, enorm, hw, S);
    k_zero<<<(2 * HDIM + 255) / 256, 256, 0, stream>>>(cs, 2 * HDIM);   // cs,cs2 contiguous
    k_bnstats<<<512, HDIM, 0, stream>>>(S, cs, cs2);
    k_bnfin<<<1, HDIM, 0, stream>>>(cs, cs2, g, be, scale, shift);
    k_apply<<<(NH4 + 255) / 256, 256, 0, stream>>>(S, scale, shift, X);
}

extern "C" void kernel_launch(void* const* d_in, const int* in_sizes, int n_in,
                              void* d_out, int out_size, void* d_ws, size_t ws_size,
                              hipStream_t stream) {
    const float* x     = (const float*)d_in[0];
    const int*   ei    = (const int*)d_in[1];
    const int*   src   = ei;
    const int*   dstp  = ei + N_EDGES;
    const int*   batch = (const int*)d_in[2];
    const float* W1 = (const float*)d_in[3];
    const float* b1 = (const float*)d_in[4];
    const float* g1 = (const float*)d_in[5];
    const float* be1= (const float*)d_in[6];
    const float* P1 = (const float*)d_in[7];
    const float* W2 = (const float*)d_in[8];
    const float* b2 = (const float*)d_in[9];
    const float* g2 = (const float*)d_in[10];
    const float* be2= (const float*)d_in[11];
    const float* W3 = (const float*)d_in[12];
    const float* b3 = (const float*)d_in[13];
    const float* g3 = (const float*)d_in[14];
    const float* be3= (const float*)d_in[15];
    const float* W4 = (const float*)d_in[16];
    const float* b4 = (const float*)d_in[17];
    const float* g4 = (const float*)d_in[18];
    const float* be4= (const float*)d_in[19];
    const float* Wl = (const float*)d_in[20];
    const float* bl = (const float*)d_in[21];
    float* out = (float*)d_out;

    // workspace carve-up (floats): 3*N*H + E + 3N + 4H + G*H + G + 512KB packed f16.
    float* ws    = (float*)d_ws;
    const size_t NH = (size_t)N_NODES * HDIM;
    float* X     = ws;                 // running features / residual (in-place)
    float* T     = ws + NH;            // hw = features @ W
    float* S     = ws + 2 * NH;        // aggregation accumulator
    float* enorm = ws + 3 * NH;
    float* deg   = enorm + N_EDGES;
    float* dis   = deg   + N_NODES;
    float* selfw = dis   + N_NODES;
    float* cs    = selfw + N_NODES;
    float* cs2   = cs    + HDIM;
    float* scale = cs2   + HDIM;
    float* shift = scale + HDIM;
    float* gsum  = shift + HDIM;
    float* gcnt  = gsum  + NGRAPH * HDIM;
    // packed f16 weights (all offsets 16B-aligned)
    _Float16* wpP1 = (_Float16*)(gcnt + NGRAPH);
    _Float16* wpW1 = wpP1 + F_INDIM * HDIM;
    _Float16* wpW2 = wpW1 + F_INDIM * HDIM;
    _Float16* wpW3 = wpW2 + HDIM * HDIM;
    _Float16* wpW4 = wpW3 + HDIM * HDIM;

    // --- gcn_norm ---
    k_zero  <<<(N_NODES + 255) / 256, 256, 0, stream>>>(deg, N_NODES);
    k_deg   <<<(N_EDGES + 255) / 256, 256, 0, stream>>>(dstp, deg);
    k_degfin<<<(N_NODES + 255) / 256, 256, 0, stream>>>(deg, dis, selfw);
    k_enorm <<<(N_EDGES + 255) / 256, 256, 0, stream>>>(src, dstp, dis, enorm);

    // --- one-time weight pre-pack to f16 fragment layout ---
    const int KH1 = F_INDIM * HDIM, KH2 = HDIM * HDIM;
    k_packW<<<(KH1 + 255) / 256, 256, 0, stream>>>(P1, wpP1, KH1);
    k_packW<<<(KH1 + 255) / 256, 256, 0, stream>>>(W1, wpW1, KH1);
    k_packW<<<(KH2 + 255) / 256, 256, 0, stream>>>(W2, wpW2, KH2);
    k_packW<<<(KH2 + 255) / 256, 256, 0, stream>>>(W3, wpW3, KH2);
    k_packW<<<(KH2 + 255) / 256, 256, 0, stream>>>(W4, wpW4, KH2);

    const int MT = N_NODES / 16;   // 6250 row tiles, exact

    // --- block 1 (128 -> 256, projection residual) ---
    k_gemm_wmma<F_INDIM><<<MT, 128, 0, stream>>>(x, wpP1, X);   // X = x @ P1
    k_gemm_wmma<F_INDIM><<<MT, 128, 0, stream>>>(x, wpW1, T);   // T = x @ W1
    run_block(stream, T, S, X, selfw, src, dstp, enorm, b1, g1, be1, cs, cs2, scale, shift);

    // --- blocks 2-4 (256 -> 256, identity residual) ---
    k_gemm_wmma<HDIM><<<MT, 128, 0, stream>>>(X, wpW2, T);
    run_block(stream, T, S, X, selfw, src, dstp, enorm, b2, g2, be2, cs, cs2, scale, shift);

    k_gemm_wmma<HDIM><<<MT, 128, 0, stream>>>(X, wpW3, T);
    run_block(stream, T, S, X, selfw, src, dstp, enorm, b3, g3, be3, cs, cs2, scale, shift);

    k_gemm_wmma<HDIM><<<MT, 128, 0, stream>>>(X, wpW4, T);
    run_block(stream, T, S, X, selfw, src, dstp, enorm, b4, g4, be4, cs, cs2, scale, shift);

    // --- global mean pool + linear head ---
    k_zero <<<(NGRAPH * HDIM + NGRAPH + 255) / 256, 256, 0, stream>>>(gsum, NGRAPH * HDIM + NGRAPH);
    k_pool <<<N_NODES, HDIM, 0, stream>>>(batch, X, gsum, gcnt);
    k_final<<<NGRAPH, HDIM, 0, stream>>>(gsum, gcnt, Wl, bl, out);
}
